// GraphProposalNetwork_52467320488000
// MI455X (gfx1250) — matrix operand order repaired and checked by reading, not compiled, not run on hardware
//
#include <hip/hip_runtime.h>
#include <hip/hip_bf16.h>

typedef __attribute__((ext_vector_type(16))) _Float16 v16h;
typedef __attribute__((ext_vector_type(8)))  float    v8f;

// lrelu(x) with slope 0.2 < 1 is exactly max(x, 0.2x). Emit v_max_num_f32
// directly: fmaxf() would make the backend insert a canonicalize (self-max)
// on the WMMA result; the raw instruction already has the wanted
// NaN-number semantics. 2 VALU ops per element, no packing moves.
__device__ __forceinline__ float lrelu(float x) {
    float y = 0.2f * x;
    float r;
    asm("v_max_num_f32 %0, %1, %2" : "=v"(r) : "v"(x), "v"(y));
    return r;
}

// One wave processes one 16-row tile of pair-rows per iteration.
// Row r -> (b = r/100, p = r%100, i = p/10, k = p%10, j = idx[i][k]).
// Layer1: pair(16x32) @ W1(32x32) via 2x wmma_f32_16x16x32_f16, bias as C
// Layer2: h(16x32) @ W2pad(32x16) via 1x wmma, bias as C
// Layer3: h2(16x8) @ W3(8x2) in VALU + argmax via shfl_xor.
// LDS tiles are wave-private; same-wave DS ops are in-order (DScnt), so no
// workgroup barriers are needed — only compiler scheduling fences.
extern "C" __global__ __launch_bounds__(256)
void gpn_wmma_kernel(const float* __restrict__ pos,
                     const float* __restrict__ ori,
                     const float* __restrict__ Wp,
                     const float* __restrict__ bp,
                     const float* __restrict__ W1,
                     const float* __restrict__ b1,
                     const float* __restrict__ W2,
                     const float* __restrict__ b2,
                     const float* __restrict__ W3,
                     const float* __restrict__ b3,
                     float* __restrict__ out,
                     int Bn, int iters, int totalTiles)
{
    __shared__ float lds[8 * 16 * 33];   // per-wave 16x33 f32 tile

    const int tid   = threadIdx.x;
    const int lane  = tid & 31;
    const int wave  = tid >> 5;
    const int n     = lane & 15;          // WMMA B/C column within tile
    const int hi    = (lane >> 4) & 1;    // which half of the wave
    const int kbA   = hi ? 8 : 0;         // A-layout K base (ISA 7.12.2)
    const int koB   = hi ? 16 : 0;        // B-layout K base (sparse-B pattern)
    const int mBase = hi ? 8 : 0;         // C-layout row base
    float* myLds = lds + wave * (16 * 33);

    // ---- Preload feat-projection weights for this lane's 8 feature dims ----
    const int d0 = hi * 8;
    float wp0[8], wp1[8], wp2[8], bpv[8];
    #pragma unroll
    for (int e = 0; e < 8; ++e) {
        wp0[e] = Wp[0 * 16 + d0 + e];
        wp1[e] = Wp[1 * 16 + d0 + e];
        wp2[e] = Wp[2 * 16 + d0 + e];
        bpv[e] = bp[d0 + e];
    }

    // ---- Preload W1 (two 16-col halves) and W2 (padded 32x16) as WMMA B ----
    v16h bw0, bw1, bw2;
    #pragma unroll
    for (int e = 0; e < 16; ++e) {
        const int K = kbA + e + ((e >= 8) ? 8 : 0);
        bw0[e] = (_Float16)W1[K * 32 + n];
        bw1[e] = (_Float16)W1[K * 32 + n + 16];
        bw2[e] = (n < 8) ? (_Float16)W2[K * 8 + n] : (_Float16)0.0f;
    }

    // ---- Biases as WMMA C operands: lane owns column n for all 8 rows ----
    const float b1a = b1[n];
    const float b1b = b1[n + 16];
    const float b2a = (n < 8) ? b2[n] : 0.0f;
    v8f cb0, cb1, cb2;
    #pragma unroll
    for (int r = 0; r < 8; ++r) { cb0[r] = b1a; cb1[r] = b1b; cb2[r] = b2a; }

    // ---- Layer-3 weights for this lane's output column (0 or 1) ----
    const int oc = hi;
    float w3v[8];
    #pragma unroll
    for (int k = 0; k < 8; ++k) w3v[k] = W3[k * 2 + oc];
    const float b3v = b3[oc];

    const int R = Bn * 100;
    const int waveGlobal = blockIdx.x * 8 + wave;
    const int waveStride = gridDim.x * 8;

    for (int it = 0; it < iters; ++it) {
        const int tile = waveGlobal + it * waveStride;
        const int row  = lane & 15;
        const int gr   = tile * 16 + row;
        const bool valid = (tile < totalTiles) && (gr < R);
        const int grc  = valid ? gr : 0;   // clamp so EXEC stays full for WMMA

        // decode pair-row -> (b, i, j)
        const int b = grc / 100;
        const int p = grc - b * 100;
        const int i = p / 10;
        const int k = p - i * 10;
        const int j = (k == 0) ? i : ((k <= i) ? (k - 1) : k);

        const float pxi = pos[(b * 10 + i) * 2 + 0];
        const float pyi = pos[(b * 10 + i) * 2 + 1];
        const float oi  = ori[b * 10 + i];
        const float pxj = pos[(b * 10 + j) * 2 + 0];
        const float pyj = pos[(b * 10 + j) * 2 + 1];
        const float oj  = ori[b * 10 + j];

        // Layer-1 A operand built directly in registers (feat_i | feat_j)
        v16h a1;
        #pragma unroll
        for (int e = 0; e < 8; ++e) {
            const float fi = fmaf(pxi, wp0[e], fmaf(pyi, wp1[e], fmaf(oi, wp2[e], bpv[e])));
            const float fj = fmaf(pxj, wp0[e], fmaf(pyj, wp1[e], fmaf(oj, wp2[e], bpv[e])));
            a1[e]     = (_Float16)fi;
            a1[e + 8] = (_Float16)fj;
        }

        // D = A*B + bias : bias rides the matrix pipe, not the VALU
        v8f acc0 = __builtin_amdgcn_wmma_f32_16x16x32_f16(false, a1, false, bw0,
                                                          (short)0, cb0, false, false);
        v8f acc1 = __builtin_amdgcn_wmma_f32_16x16x32_f16(false, a1, false, bw1,
                                                          (short)0, cb1, false, false);

        // lrelu, stage h1 (16x32) in LDS for the layout transpose
        #pragma unroll
        for (int r = 0; r < 8; ++r) {
            myLds[(r + mBase) * 33 + n]      = lrelu(acc0[r]);
            myLds[(r + mBase) * 33 + n + 16] = lrelu(acc1[r]);
        }
        __builtin_amdgcn_sched_barrier(0);   // keep DS stores before DS loads

        // Layer-2 A operand gathered from LDS in ISA A-layout
        v16h a2;
        #pragma unroll
        for (int e = 0; e < 16; ++e) {
            const int K = kbA + e + ((e >= 8) ? 8 : 0);
            a2[e] = (_Float16)myLds[row * 33 + K];
        }

        v8f acc2 = __builtin_amdgcn_wmma_f32_16x16x32_f16(false, a2, false, bw2,
                                                          (short)0, cb2, false, false);

        // lrelu, stage h2 (16x8, cols>=8 are don't-care)
        #pragma unroll
        for (int r = 0; r < 8; ++r) {
            myLds[(r + mBase) * 33 + n] = lrelu(acc2[r]);
        }
        __builtin_amdgcn_sched_barrier(0);   // keep DS stores before DS loads

        // Layer 3: each lane = (row, output column); dot over 8 dims
        float ev = b3v;
        #pragma unroll
        for (int kk = 0; kk < 8; ++kk)
            ev = fmaf(myLds[row * 33 + kk], w3v[kk], ev);
        const float evOther = __shfl_xor(ev, 16, 32);

        if (valid) {
            // streaming outputs, never re-read: non-temporal stores
            __builtin_nontemporal_store(ev, &out[(size_t)gr * 2 + oc]);
            if (oc == 0) {
                // argmax over axis of size 2: index 1 iff logit1 > logit0
                __builtin_nontemporal_store((evOther > ev) ? 1.0f : 0.0f,
                                            &out[(size_t)R * 2 + gr]);
            }
        }
    }
}

extern "C" void kernel_launch(void* const* d_in, const int* in_sizes, int n_in,
                              void* d_out, int out_size, void* d_ws, size_t ws_size,
                              hipStream_t stream) {
    (void)n_in; (void)out_size; (void)d_ws; (void)ws_size;
    const float* pos = (const float*)d_in[0];
    const float* ori = (const float*)d_in[1];
    const float* Wp  = (const float*)d_in[2];
    const float* bp  = (const float*)d_in[3];
    const float* W1  = (const float*)d_in[4];
    const float* b1  = (const float*)d_in[5];
    const float* W2  = (const float*)d_in[6];
    const float* b2  = (const float*)d_in[7];
    const float* W3  = (const float*)d_in[8];
    const float* b3  = (const float*)d_in[9];

    const int Bn = in_sizes[0] / 20;              // (B, N=10, 2)
    const int R  = Bn * 100;
    const int totalTiles = (R + 15) / 16;         // 102,400 for B=16384

    int blocks = 3200;                            // 25,600 waves, 8 per block
    const int wavesTotal = blocks * 8;
    const int iters = (totalTiles + wavesTotal - 1) / wavesTotal;  // 4

    gpn_wmma_kernel<<<blocks, 256, 0, stream>>>(pos, ori, Wp, bp, W1, b1,
                                                W2, b2, W3, b3,
                                                (float*)d_out, Bn, iters, totalTiles);
}
